// WindowAttention_59184649339031
// MI455X (gfx1250) — matrix operand order repaired and checked by reading, not compiled, not run on hardware
//
#include <hip/hip_runtime.h>
#include <hip/hip_bf16.h>

typedef _Float16 f16;
typedef __attribute__((ext_vector_type(16))) _Float16 v16h;
typedef __attribute__((ext_vector_type(4)))  _Float16 v4h;
typedef __attribute__((ext_vector_type(8)))  float    v8f;
typedef __attribute__((ext_vector_type(4)))  float    v4f;

// D = A(16x32 f16) * B(32x16 f16) + C(16x16 f32), wave32
__device__ __forceinline__ v8f wmma16x16x32(v16h a, v16h b, v8f c) {
  return __builtin_amdgcn_wmma_f32_16x16x32_f16(
      /*neg_a=*/false, a, /*neg_b=*/false, b,
      /*c_mod=*/(short)0, c, /*reuse_a=*/false, /*reuse_b=*/false);
}

// A operand: 16(M) x 32(K), row-major source (row_stride in f16 elems).
// ISA layout: lanes 0-15 -> M=lane, K chunks {0..7, 16..23};
//             lanes 16-31 -> M=lane-16, K chunks {8..15, 24..31}.
__device__ __forceinline__ v16h load_a_frag(const f16* __restrict__ base,
                                            int row_stride, int lane) {
  const int l  = lane & 15;
  const int kb = (lane & 16) ? 8 : 0;
  const f16* p = base + (size_t)l * row_stride + kb;
  union { v16h v; v4f q[2]; } u;
  u.q[0] = *(const v4f*)(p);        // K = kb .. kb+7
  u.q[1] = *(const v4f*)(p + 16);   // K = kb+16 .. kb+23
  return u.v;
}

// B operand: 32(K) x 16(N); source stores each column's 32 K-values
// contiguously (col_stride f16 elems between columns).
// ISA layout: lanes 0-15 -> col=lane, K chunks {0..7, 16..23}; lanes 16-31 -> col=lane-16, K chunks {8..15, 24..31}.
__device__ __forceinline__ v16h load_b_frag(const f16* __restrict__ base,
                                            int col_stride, int lane) {
  const int c  = lane & 15;
  const int kb = (lane & 16) ? 8 : 0;
  const f16* p = base + (size_t)c * col_stride + kb;
  union { v16h v; v4f q[2]; } u;
  u.q[0] = *(const v4f*)(p);
  u.q[1] = *(const v4f*)(p + 16);
  return u.v;
}

// ---------------- fp32 -> fp16 conversion (vectorized) ----------------
__global__ void cvt_f32_to_f16_x4(const float* __restrict__ src,
                                  f16* __restrict__ dst, size_t n4) {
  size_t i = (size_t)blockIdx.x * blockDim.x + threadIdx.x;
  const size_t stride = (size_t)gridDim.x * blockDim.x;
  for (; i < n4; i += stride) {
    v4f v = ((const v4f*)src)[i];
    v4h h;
    h.x = (f16)v.x; h.y = (f16)v.y; h.z = (f16)v.z; h.w = (f16)v.w;
    *(v4h*)(dst + 4 * i) = h;
  }
}

// ---------------- fused QKV projection GEMM ----------------
// Y(262144 x 768) = x16 @ [wq|wkv]^T + bias, written as q/k/v in (B_,H,N,hd) f16.
// Block: 128 thr = 4 waves; block tile 64(M) x 64(N); wave tile 16 x 64.
__global__ __launch_bounds__(128)
void qkv_gemm(const f16* __restrict__ x16,
              const f16* __restrict__ wq16, const f16* __restrict__ wkv16,
              const float* __restrict__ bq, const float* __restrict__ bkv,
              f16* __restrict__ q16, f16* __restrict__ k16, f16* __restrict__ v16) {
  const int lane  = threadIdx.x & 31;
  const int wave  = threadIdx.x >> 5;
  const int mbase = blockIdx.x * 64 + wave * 16;
  const int nbase = blockIdx.y * 64;

  v8f acc[4];
  #pragma unroll
  for (int t = 0; t < 4; ++t) {
    const int jcol = nbase + t * 16 + (lane & 15);
    const float bv = (jcol < 256) ? bq[jcol] : bkv[jcol - 256];
    #pragma unroll
    for (int r = 0; r < 8; ++r) acc[t][r] = bv;
  }

  #pragma unroll
  for (int k0 = 0; k0 < 256; k0 += 32) {
    const v16h a = load_a_frag(x16 + (size_t)mbase * 256 + k0, 256, lane);
    #pragma unroll
    for (int t = 0; t < 4; ++t) {
      const int jt = nbase + t * 16;
      const f16* wbase = (jt < 256) ? (wq16 + (size_t)jt * 256)
                                    : (wkv16 + (size_t)(jt - 256) * 256);
      const v16h b = load_b_frag(wbase + k0, 256, lane);
      acc[t] = wmma16x16x32(a, b, acc[t]);
    }
  }

  const int hi = (lane & 16) ? 8 : 0;
  #pragma unroll
  for (int t = 0; t < 4; ++t) {
    const int jcol = nbase + t * 16 + (lane & 15);
    f16* dst; int jj; float scale;
    if (jcol < 256)      { dst = q16; jj = jcol;       scale = 0.17677669529663687f; } // hd^-0.5
    else if (jcol < 512) { dst = k16; jj = jcol - 256; scale = 1.0f; }
    else                 { dst = v16; jj = jcol - 512; scale = 1.0f; }
    const int h = jj >> 5, d = jj & 31;
    #pragma unroll
    for (int r = 0; r < 8; ++r) {
      const int m = mbase + r + hi;
      const int b = m >> 6, n = m & 63;
      dst[(((size_t)b * 8 + h) * 64 + n) * 32 + d] = (f16)(acc[t][r] * scale);
    }
  }
}

// ---------------- fused window attention ----------------
// One block per (b,h). 128 thr = 4 waves; wave handles a 16-row strip of the
// 64x64 score matrix. hd=32 => one WMMA K-step for QK^T; K=64 => two for PV.
__global__ __launch_bounds__(128)
void win_attn(const f16* __restrict__ q16, const f16* __restrict__ k16,
              const f16* __restrict__ v16, const float* __restrict__ mask,
              const float* __restrict__ bias_table, f16* __restrict__ ctx16) {
  __shared__ __align__(16) float sbm[64 * 64];     // rel-pos bias + shift mask
  __shared__ __align__(16) f16   probs[64 * 72];   // softmax probs, padded stride
  __shared__ __align__(16) f16   vT[32 * 72];      // V transposed: [d][key]

  const int bh = blockIdx.x;
  const int b  = bh >> 3;
  const int h  = bh & 7;
  const int w  = b & 63;           // nW = 64
  const int tid  = threadIdx.x;
  const int lane = tid & 31;
  const int wave = tid >> 5;

  // Precompute bias+mask table for this (h, window)
  for (int e = tid; e < 4096; e += 128) {
    const int nq = e >> 6, m = e & 63;
    const int dy = (nq >> 3) - (m >> 3) + 7;
    const int dx = (nq & 7) - (m & 7) + 7;
    sbm[e] = bias_table[(dy * 15 + dx) * 8 + h] +
             mask[((size_t)w * 64 + nq) * 64 + m];
  }

  // Stage V transposed: vT[d*72 + key]
  {
    const int n    = tid >> 1;
    const int dseg = (tid & 1) * 16;
    union { v4f q[2]; f16 e[16]; } u;
    const f16* src = v16 + (size_t)bh * 2048 + (size_t)n * 32 + dseg;
    u.q[0] = *(const v4f*)(src);
    u.q[1] = *(const v4f*)(src + 8);
    #pragma unroll
    for (int i = 0; i < 16; ++i) vT[(dseg + i) * 72 + n] = u.e[i];
  }
  __syncthreads();

  const int strip = wave * 16;
  const int hi  = (lane & 16) ? 8 : 0;
  const int col = lane & 15;

  // S = (q*scale) @ k^T  -- A frags direct from global (row-major, stride 32)
  const v16h aq = load_a_frag(q16 + (size_t)bh * 2048 + (size_t)strip * 32, 32, lane);
  v8f s[4];
  #pragma unroll
  for (int t = 0; t < 4; ++t) {
    const v16h bk = load_b_frag(k16 + (size_t)bh * 2048 + (size_t)(t * 16) * 32, 32, lane);
    v8f z = {};
    s[t] = wmma16x16x32(aq, bk, z);
  }

  // + rel-pos bias + mask
  #pragma unroll
  for (int t = 0; t < 4; ++t)
    #pragma unroll
    for (int r = 0; r < 8; ++r)
      s[t][r] += sbm[(strip + r + hi) * 64 + t * 16 + col];

  // Row softmax: 64 cols live across 4 accumulators x 16 lanes of a half-wave.
  #pragma unroll
  for (int r = 0; r < 8; ++r) {
    const int nq = strip + r + hi;
    float mx = fmaxf(fmaxf(s[0][r], s[1][r]), fmaxf(s[2][r], s[3][r]));
    #pragma unroll
    for (int off = 1; off < 16; off <<= 1)
      mx = fmaxf(mx, __shfl_xor(mx, off, 32));
    const float e0 = __expf(s[0][r] - mx), e1 = __expf(s[1][r] - mx);
    const float e2 = __expf(s[2][r] - mx), e3 = __expf(s[3][r] - mx);
    float sum = e0 + e1 + e2 + e3;
    #pragma unroll
    for (int off = 1; off < 16; off <<= 1)
      sum += __shfl_xor(sum, off, 32);
    const float inv = 1.0f / sum;
    probs[nq * 72 +  0 + col] = (f16)(e0 * inv);
    probs[nq * 72 + 16 + col] = (f16)(e1 * inv);
    probs[nq * 72 + 32 + col] = (f16)(e2 * inv);
    probs[nq * 72 + 48 + col] = (f16)(e3 * inv);
  }
  // Same-wave DS ops are in-order; no barrier needed (strip is wave-private).

  // O = P @ V : A from probs (LDS), B from vT (LDS)
  v8f o[2] = { {}, {} };
  #pragma unroll
  for (int ks = 0; ks < 64; ks += 32) {
    const v16h ap = load_a_frag(probs + strip * 72 + ks, 72, lane);
    #pragma unroll
    for (int tn = 0; tn < 2; ++tn) {
      const v16h bv = load_b_frag(vT + (tn * 16) * 72 + ks, 72, lane);
      o[tn] = wmma16x16x32(ap, bv, o[tn]);
    }
  }

  // ctx in (B_, N, C) f16
  #pragma unroll
  for (int tn = 0; tn < 2; ++tn)
    #pragma unroll
    for (int r = 0; r < 8; ++r) {
      const int nq = strip + r + hi;
      const int c  = h * 32 + tn * 16 + col;
      ctx16[((size_t)b * 64 + nq) * 256 + c] = (f16)o[tn][r];
    }
}

// ---------------- output projection GEMM ----------------
__global__ __launch_bounds__(128)
void proj_gemm(const f16* __restrict__ ctx16, const f16* __restrict__ wp16,
               const float* __restrict__ bproj, float* __restrict__ out) {
  const int lane  = threadIdx.x & 31;
  const int wave  = threadIdx.x >> 5;
  const int mbase = blockIdx.x * 64 + wave * 16;
  const int nbase = blockIdx.y * 64;

  v8f acc[4];
  #pragma unroll
  for (int t = 0; t < 4; ++t) {
    const float bv = bproj[nbase + t * 16 + (lane & 15)];
    #pragma unroll
    for (int r = 0; r < 8; ++r) acc[t][r] = bv;
  }

  #pragma unroll
  for (int k0 = 0; k0 < 256; k0 += 32) {
    const v16h a = load_a_frag(ctx16 + (size_t)mbase * 256 + k0, 256, lane);
    #pragma unroll
    for (int t = 0; t < 4; ++t) {
      const v16h bw = load_b_frag(wp16 + (size_t)(nbase + t * 16) * 256 + k0, 256, lane);
      acc[t] = wmma16x16x32(a, bw, acc[t]);
    }
  }

  const int hi = (lane & 16) ? 8 : 0;
  #pragma unroll
  for (int t = 0; t < 4; ++t) {
    const int jcol = nbase + t * 16 + (lane & 15);
    #pragma unroll
    for (int r = 0; r < 8; ++r) {
      const int m = mbase + r + hi;
      out[(size_t)m * 256 + jcol] = acc[t][r];
    }
  }
}

extern "C" void kernel_launch(void* const* d_in, const int* in_sizes, int n_in,
                              void* d_out, int out_size, void* d_ws, size_t ws_size,
                              hipStream_t stream) {
  (void)in_sizes; (void)n_in; (void)out_size; (void)ws_size;
  const float* x    = (const float*)d_in[0];
  const float* mask = (const float*)d_in[1];
  const float* wq   = (const float*)d_in[2];
  const float* bq   = (const float*)d_in[3];
  const float* wkv  = (const float*)d_in[4];
  const float* bkv  = (const float*)d_in[5];
  const float* bt   = (const float*)d_in[6];
  const float* wp   = (const float*)d_in[7];
  const float* bp   = (const float*)d_in[8];
  float* out = (float*)d_out;

  const size_t X = 4096ull * 64 * 256;   // 67,108,864 elements
  f16* base  = (f16*)d_ws;
  f16* x16   = base;
  f16* q16   = base + X;
  f16* k16   = base + 2 * X;
  f16* v16   = base + 3 * X;
  f16* ctx16 = base + 4 * X;
  f16* wq16  = base + 5 * X;
  f16* wkv16 = wq16 + 256 * 256;
  f16* wp16  = wkv16 + 512 * 256;

  cvt_f32_to_f16_x4<<<8192, 256, 0, stream>>>(x,  x16,  X / 4);
  cvt_f32_to_f16_x4<<<64,   256, 0, stream>>>(wq, wq16, (256 * 256) / 4);
  cvt_f32_to_f16_x4<<<128,  256, 0, stream>>>(wkv, wkv16, (512 * 256) / 4);
  cvt_f32_to_f16_x4<<<64,   256, 0, stream>>>(wp, wp16, (256 * 256) / 4);

  qkv_gemm<<<dim3(4096, 12), 128, 0, stream>>>(x16, wq16, wkv16, bq, bkv,
                                               q16, k16, v16);
  win_attn<<<32768, 128, 0, stream>>>(q16, k16, v16, mask, bt, ctx16);
  proj_gemm<<<dim3(4096, 4), 128, 0, stream>>>(ctx16, wp16, bp, out);
}
